// CausalSelfFlexBlock_34084860461624
// MI455X (gfx1250) — matrix operand order repaired and checked by reading, not compiled, not run on hardware
//
#include <hip/hip_runtime.h>

// ---------------------------------------------------------------------------
// Types / helpers for CDNA5 (gfx1250) WMMA bf16 path
// ---------------------------------------------------------------------------
typedef unsigned short u16;
typedef __attribute__((ext_vector_type(16))) __bf16 bf16x16;
typedef __attribute__((ext_vector_type(8)))  float  f32x8;

union FragAB {
  bf16x16 v;
  uint4   q[2];
  u16     u[16];
};

__device__ __forceinline__ u16 f2bf(float f) {
  unsigned u = __float_as_uint(f);
  u += 0x7fffu + ((u >> 16) & 1u);   // round-to-nearest-even
  return (u16)(u >> 16);
}
__device__ __forceinline__ float bf2f(u16 h) {
  return __uint_as_float(((unsigned)h) << 16);
}
__device__ __forceinline__ f32x8 wmma_bf16(bf16x16 a, bf16x16 b, f32x8 c) {
  // D = A(16x32) * B(32x16) + C, f32 accumulate
  return __builtin_amdgcn_wmma_f32_16x16x32_bf16(false, a, false, b, (short)0, c,
                                                 false, false);
}

// Async global->LDS copy (16B per lane), tracked by ASYNCcnt.
__device__ __forceinline__ void async_copy_b128(u16* lds_dst, const u16* gsrc) {
  auto l = (__attribute__((address_space(3))) u16*)lds_dst;
  unsigned laddr = (unsigned)(unsigned long long)l;
  asm volatile("global_load_async_to_lds_b128 %0, %1, off"
               :: "v"(laddr), "v"(gsrc) : "memory");
}
__device__ __forceinline__ void wait_async0() {
  asm volatile("s_wait_asynccnt 0x0" ::: "memory");
}

#define D_MODEL 1024
#define N_HEADS 16
#define HEAD_DIM 64
#define D_FF 4096
#define SEQ_L 2048
#define BATCH 2
#define ROWS (BATCH * SEQ_L)   // 4096

// ---------------------------------------------------------------------------
// f32 -> bf16 conversion (weights)
// ---------------------------------------------------------------------------
__global__ __launch_bounds__(256)
void f32_to_bf16_kernel(const float* __restrict__ src, u16* __restrict__ dst, int n) {
  int i = blockIdx.x * 256 + threadIdx.x;
  if (i < n) dst[i] = f2bf(src[i]);
}

// ---------------------------------------------------------------------------
// LayerNorm over D_MODEL, output bf16.  One block (8 waves) per row.
// ---------------------------------------------------------------------------
__global__ __launch_bounds__(256)
void ln_bf16_kernel(const float* __restrict__ x, const float* __restrict__ g,
                    const float* __restrict__ bta, u16* __restrict__ out) {
  __shared__ float red[8];
  const int t = threadIdx.x;
  const int wave = t >> 5, lane = t & 31;
  const float* xr = x + (size_t)blockIdx.x * D_MODEL;

  float v[4];
  float s = 0.f;
#pragma unroll
  for (int i = 0; i < 4; ++i) { v[i] = xr[t + 256 * i]; s += v[i]; }
#pragma unroll
  for (int o = 16; o >= 1; o >>= 1) s += __shfl_xor(s, o);
  if (lane == 0) red[wave] = s;
  __syncthreads();
  if (t == 0) {
    float tt = 0.f;
    for (int i = 0; i < 8; ++i) tt += red[i];
    red[0] = tt;
  }
  __syncthreads();
  const float mu = red[0] * (1.f / D_MODEL);
  __syncthreads();

  float vs = 0.f;
#pragma unroll
  for (int i = 0; i < 4; ++i) { float d = v[i] - mu; vs += d * d; }
#pragma unroll
  for (int o = 16; o >= 1; o >>= 1) vs += __shfl_xor(vs, o);
  if (lane == 0) red[wave] = vs;
  __syncthreads();
  if (t == 0) {
    float tt = 0.f;
    for (int i = 0; i < 8; ++i) tt += red[i];
    red[0] = tt;
  }
  __syncthreads();
  const float rst = rsqrtf(red[0] * (1.f / D_MODEL) + 1e-5f);

  u16* orow = out + (size_t)blockIdx.x * D_MODEL;
#pragma unroll
  for (int i = 0; i < 4; ++i) {
    const int c = t + 256 * i;
    orow[c] = f2bf((v[i] - mu) * rst * g[c] + bta[c]);
  }
}

// ---------------------------------------------------------------------------
// Generic GEMM:  out[M,N] = A[M,K](bf16) @ W[N,K]^T(bf16) (+bias) (+residual)
// 256 threads = 8 waves; block tile 128x64; wave tile 32x32 (4 accumulators,
// 2x fragment reuse); K-step 32.  Staging via global_load_async_to_lds_b128.
// ---------------------------------------------------------------------------
__global__ __launch_bounds__(256)
void gemm_bf16_kernel(const u16* __restrict__ A, const u16* __restrict__ W,
                      const float* __restrict__ bias, const float* __restrict__ resid,
                      float* __restrict__ outF, u16* __restrict__ outB,
                      int M, int N, int K) {
  __shared__ __align__(16) u16 lA[128 * 32];   // 8 KiB
  __shared__ __align__(16) u16 lB[64 * 32];    // 4 KiB
  const int t = threadIdx.x;
  const int wave = t >> 5, lane = t & 31;
  const int half = lane >> 4, nl = lane & 15;
  const int wr = wave & 3, wc = wave >> 2;     // 4 M-subtiles x 2 N-subtiles
  const int m0 = blockIdx.y * 128, n0 = blockIdx.x * 64;
  // lA staging: 2 x b128 per thread (row t/2, col (t&1)*16 + {0,8})
  const int arow = t >> 1, acol = (t & 1) * 16;
  // lB staging: 1 x b128 per thread (row t/4, col (t&3)*8)
  const int brow = t >> 2, bcol = (t & 3) * 8;
  (void)M;

  f32x8 acc00, acc01, acc10, acc11;
#pragma unroll
  for (int r = 0; r < 8; ++r) {
    acc00[r] = 0.f; acc01[r] = 0.f; acc10[r] = 0.f; acc11[r] = 0.f;
  }

  for (int k0 = 0; k0 < K; k0 += 32) {
    async_copy_b128(&lA[arow * 32 + acol],     &A[(size_t)(m0 + arow) * K + k0 + acol]);
    async_copy_b128(&lA[arow * 32 + acol + 8], &A[(size_t)(m0 + arow) * K + k0 + acol + 8]);
    async_copy_b128(&lB[brow * 32 + bcol],     &W[(size_t)(n0 + brow) * K + k0 + bcol]);
    wait_async0();
    __syncthreads();
    if (k0 + 32 < K) {   // global_prefetch_b8 of next tiles
      __builtin_prefetch(&A[(size_t)(m0 + arow) * K + k0 + 32 + acol], 0, 0);
      __builtin_prefetch(&W[(size_t)(n0 + brow) * K + k0 + 32 + bcol], 0, 0);
    }

    const int klo = half ? 16 : 0;
    FragAB fa0, fa1;
    {
      const int ar = 32 * wr + nl;
      fa0.q[0] = *(const uint4*)&lA[ar * 32 + klo];
      fa0.q[1] = *(const uint4*)&lA[ar * 32 + 8 + klo];
      fa1.q[0] = *(const uint4*)&lA[(ar + 16) * 32 + klo];
      fa1.q[1] = *(const uint4*)&lA[(ar + 16) * 32 + 8 + klo];
    }
    FragAB fb0, fb1;
    {
      const int bk = half ? 16 : 0;
      const int br = 32 * wc + nl;
      fb0.q[0] = *(const uint4*)&lB[br * 32 + bk];
      fb0.q[1] = *(const uint4*)&lB[br * 32 + bk + 8];
      fb1.q[0] = *(const uint4*)&lB[(br + 16) * 32 + bk];
      fb1.q[1] = *(const uint4*)&lB[(br + 16) * 32 + bk + 8];
    }
    acc00 = wmma_bf16(fa0.v, fb0.v, acc00);
    acc01 = wmma_bf16(fa0.v, fb1.v, acc01);
    acc10 = wmma_bf16(fa1.v, fb0.v, acc10);
    acc11 = wmma_bf16(fa1.v, fb1.v, acc11);
    __syncthreads();
  }

#pragma unroll
  for (int i = 0; i < 2; ++i) {
#pragma unroll
    for (int j = 0; j < 2; ++j) {
      const f32x8 av = i ? (j ? acc11 : acc10) : (j ? acc01 : acc00);
      const int ncol = n0 + 32 * wc + 16 * j + nl;
      const float bi = bias ? bias[ncol] : 0.f;
#pragma unroll
      for (int r = 0; r < 8; ++r) {
        const int mrow = m0 + 32 * wr + 16 * i + r + 8 * half;
        float o = av[r] + bi;
        if (resid) o += resid[(size_t)mrow * N + ncol];
        if (outF) outF[(size_t)mrow * N + ncol] = o;
        else      outB[(size_t)mrow * N + ncol] = f2bf(o);
      }
    }
  }
}

// ---------------------------------------------------------------------------
// Fused SwiGLU gate GEMM: out = silu(A@W1^T) * (A@W2^T)  (bf16 out)
// Block tile 128x64, wave tile 32x32 against BOTH weight matrices
// (8 WMMAs / K-step / wave).  Async LDS staging.
// ---------------------------------------------------------------------------
__global__ __launch_bounds__(256)
void mlp_gate_kernel(const u16* __restrict__ A, const u16* __restrict__ W1,
                     const u16* __restrict__ W2, u16* __restrict__ outB,
                     int M, int N, int K) {
  __shared__ __align__(16) u16 lA[128 * 32];   // 8 KiB
  __shared__ __align__(16) u16 l1[64 * 32];    // 4 KiB
  __shared__ __align__(16) u16 l2[64 * 32];    // 4 KiB
  const int t = threadIdx.x;
  const int wave = t >> 5, lane = t & 31;
  const int half = lane >> 4, nl = lane & 15;
  const int wr = wave & 3, wc = wave >> 2;
  const int m0 = blockIdx.y * 128, n0 = blockIdx.x * 64;
  const int arow = t >> 1, acol = (t & 1) * 16;
  const int brow = t >> 2, bcol = (t & 3) * 8;
  (void)M;

  f32x8 g00, g01, g10, g11, u00, u01, u10, u11;
#pragma unroll
  for (int r = 0; r < 8; ++r) {
    g00[r] = 0.f; g01[r] = 0.f; g10[r] = 0.f; g11[r] = 0.f;
    u00[r] = 0.f; u01[r] = 0.f; u10[r] = 0.f; u11[r] = 0.f;
  }

  for (int k0 = 0; k0 < K; k0 += 32) {
    async_copy_b128(&lA[arow * 32 + acol],     &A[(size_t)(m0 + arow) * K + k0 + acol]);
    async_copy_b128(&lA[arow * 32 + acol + 8], &A[(size_t)(m0 + arow) * K + k0 + acol + 8]);
    async_copy_b128(&l1[brow * 32 + bcol],     &W1[(size_t)(n0 + brow) * K + k0 + bcol]);
    async_copy_b128(&l2[brow * 32 + bcol],     &W2[(size_t)(n0 + brow) * K + k0 + bcol]);
    wait_async0();
    __syncthreads();
    if (k0 + 32 < K) {
      __builtin_prefetch(&A[(size_t)(m0 + arow) * K + k0 + 32 + acol], 0, 0);
      __builtin_prefetch(&W1[(size_t)(n0 + brow) * K + k0 + 32 + bcol], 0, 0);
      __builtin_prefetch(&W2[(size_t)(n0 + brow) * K + k0 + 32 + bcol], 0, 0);
    }

    const int klo = half ? 16 : 0;
    FragAB fa0, fa1;
    {
      const int ar = 32 * wr + nl;
      fa0.q[0] = *(const uint4*)&lA[ar * 32 + klo];
      fa0.q[1] = *(const uint4*)&lA[ar * 32 + 8 + klo];
      fa1.q[0] = *(const uint4*)&lA[(ar + 16) * 32 + klo];
      fa1.q[1] = *(const uint4*)&lA[(ar + 16) * 32 + 8 + klo];
    }
    const int bk = half ? 16 : 0;
    const int br0 = 32 * wc + nl;
    const int br1 = br0 + 16;
    FragAB f10, f11, f20, f21;
    f10.q[0] = *(const uint4*)&l1[br0 * 32 + bk];
    f10.q[1] = *(const uint4*)&l1[br0 * 32 + bk + 8];
    f11.q[0] = *(const uint4*)&l1[br1 * 32 + bk];
    f11.q[1] = *(const uint4*)&l1[br1 * 32 + bk + 8];
    f20.q[0] = *(const uint4*)&l2[br0 * 32 + bk];
    f20.q[1] = *(const uint4*)&l2[br0 * 32 + bk + 8];
    f21.q[0] = *(const uint4*)&l2[br1 * 32 + bk];
    f21.q[1] = *(const uint4*)&l2[br1 * 32 + bk + 8];

    g00 = wmma_bf16(fa0.v, f10.v, g00);
    g01 = wmma_bf16(fa0.v, f11.v, g01);
    g10 = wmma_bf16(fa1.v, f10.v, g10);
    g11 = wmma_bf16(fa1.v, f11.v, g11);
    u00 = wmma_bf16(fa0.v, f20.v, u00);
    u01 = wmma_bf16(fa0.v, f21.v, u01);
    u10 = wmma_bf16(fa1.v, f20.v, u10);
    u11 = wmma_bf16(fa1.v, f21.v, u11);
    __syncthreads();
  }

#pragma unroll
  for (int i = 0; i < 2; ++i) {
#pragma unroll
    for (int j = 0; j < 2; ++j) {
      const f32x8 gv = i ? (j ? g11 : g10) : (j ? g01 : g00);
      const f32x8 uv = i ? (j ? u11 : u10) : (j ? u01 : u00);
      const int ncol = n0 + 32 * wc + 16 * j + nl;
#pragma unroll
      for (int r = 0; r < 8; ++r) {
        const int mrow = m0 + 32 * wr + 16 * i + r + 8 * half;
        const float g = gv[r];
        const float sg = g / (1.f + __expf(-g));     // silu
        outB[(size_t)mrow * N + ncol] = f2bf(sg * uv[r]);
      }
    }
  }
}

// ---------------------------------------------------------------------------
// RoPE + head split: qkv[B*L,3072](bf16) -> q,k,v [B,H,L,64](bf16)
// ---------------------------------------------------------------------------
__global__ __launch_bounds__(256)
void rope_split_kernel(const u16* __restrict__ qkv, u16* __restrict__ q,
                       u16* __restrict__ k, u16* __restrict__ v) {
  const int idx = blockIdx.x * 256 + threadIdx.x;   // < 2*2048*16*32 = 2^21
  const int i   = idx & 31;
  const int h   = (idx >> 5) & 15;
  const int pos = (idx >> 9) & 2047;
  const int b   = idx >> 20;

  const float ang = (float)pos * __powf(10000.f, -(float)(2 * i) * (1.f / 64.f));
  const float c = __cosf(ang), s = __sinf(ang);
  const size_t row = (size_t)(b * SEQ_L + pos) * (3 * D_MODEL);
  const size_t ob  = ((size_t)(b * N_HEADS + h) * SEQ_L + pos) * HEAD_DIM;

  {
    const float x1 = bf2f(qkv[row + h * 64 + i]);
    const float x2 = bf2f(qkv[row + h * 64 + i + 32]);
    q[ob + i]      = f2bf(x1 * c - x2 * s);
    q[ob + i + 32] = f2bf(x2 * c + x1 * s);
  }
  {
    const float x1 = bf2f(qkv[row + D_MODEL + h * 64 + i]);
    const float x2 = bf2f(qkv[row + D_MODEL + h * 64 + i + 32]);
    k[ob + i]      = f2bf(x1 * c - x2 * s);
    k[ob + i + 32] = f2bf(x2 * c + x1 * s);
  }
  v[ob + i]      = qkv[row + 2 * D_MODEL + h * 64 + i];
  v[ob + i + 32] = qkv[row + 2 * D_MODEL + h * 64 + i + 32];
}

// ---------------------------------------------------------------------------
// Flash attention: one wave owns one (b,h,16-row Q tile). Causal early exit.
// Scores & P*V via WMMA bf16. Online softmax, stats duplicated per 16-lane half.
// ---------------------------------------------------------------------------
__global__ __launch_bounds__(256)
void attn_kernel(const u16* __restrict__ Q, const u16* __restrict__ Kt,
                 const u16* __restrict__ V, const unsigned char* __restrict__ kpm,
                 u16* __restrict__ outp) {
  __shared__ __align__(16) u16 sP[8][16 * 32];     // per-wave P tile (16 x 32)
  __shared__ __align__(16) u16 sVt[8][64 * 32];    // per-wave V^T tile (64 x 32)
  const int tid = threadIdx.x;
  const int wave = tid >> 5, lane = tid & 31;
  const int half = lane >> 4, nl = lane & 15;
  const int gw = blockIdx.x * 8 + wave;            // 0..4095
  const int qt = gw & 127;
  const int h  = (gw >> 7) & 15;
  const int b  = gw >> 11;
  const int q0 = qt * 16;

  const u16* Qb = Q  + (size_t)(b * N_HEADS + h) * (SEQ_L * HEAD_DIM);
  const u16* Kb = Kt + (size_t)(b * N_HEADS + h) * (SEQ_L * HEAD_DIM);
  const u16* Vb = V  + (size_t)(b * N_HEADS + h) * (SEQ_L * HEAD_DIM);
  const unsigned char* pm = kpm + (size_t)b * SEQ_L;

  FragAB fq[2];
  {
    const int qr = q0 + nl;
    const int klo = half ? 16 : 0;
#pragma unroll
    for (int dc = 0; dc < 2; ++dc) {
      fq[dc].q[0] = *(const uint4*)&Qb[(size_t)qr * 64 + 32 * dc + klo];
      fq[dc].q[1] = *(const uint4*)&Qb[(size_t)qr * 64 + 32 * dc + 8 + klo];
    }
  }

  f32x8 acc[4];
  float rmax[8], rsum[8];
#pragma unroll
  for (int r = 0; r < 8; ++r) {
    rmax[r] = -1e30f; rsum[r] = 0.f;
    acc[0][r] = 0.f; acc[1][r] = 0.f; acc[2][r] = 0.f; acc[3][r] = 0.f;
  }

  const int kend = q0 + 16;                        // causal early exit
  for (int k0 = 0; k0 < kend; k0 += 32) {
    // ---- S = Q K^T : two 16(q) x 16(key) tiles ----
    f32x8 s[2];
#pragma unroll
    for (int r = 0; r < 8; ++r) { s[0][r] = 0.f; s[1][r] = 0.f; }
#pragma unroll
    for (int j = 0; j < 2; ++j) {
      const int key = k0 + 16 * j + nl;
      const int kc  = key < SEQ_L ? key : SEQ_L - 1;
#pragma unroll
      for (int dc = 0; dc < 2; ++dc) {
        FragAB fk;
        const int dof = 32 * dc + (half ? 16 : 0);
        fk.q[0] = *(const uint4*)&Kb[(size_t)kc * 64 + dof];
        fk.q[1] = *(const uint4*)&Kb[(size_t)kc * 64 + dof + 8];
        s[j] = wmma_bf16(fq[dc].v, fk.v, s[j]);
      }
    }

    // ---- mask (causal + key padding) and scale by 1/sqrt(64) ----
#pragma unroll
    for (int j = 0; j < 2; ++j) {
#pragma unroll
      for (int r = 0; r < 8; ++r) {
        const int m = r + 8 * half;
        const int key = k0 + 16 * j + nl;
        const int kc = key < SEQ_L ? key : SEQ_L - 1;
        const bool keep = (key <= q0 + m) && (key < SEQ_L) && (pm[kc] == 0);
        s[j][r] = keep ? s[j][r] * 0.125f : -1e30f;
      }
    }

    // ---- online softmax update (per-row stats live in each 16-lane half) ----
#pragma unroll
    for (int r = 0; r < 8; ++r) {
      float tm = fmaxf(s[0][r], s[1][r]);
#pragma unroll
      for (int o = 8; o >= 1; o >>= 1) tm = fmaxf(tm, __shfl_xor(tm, o));
      const float nm = fmaxf(rmax[r], tm);
      const float corr = __expf(rmax[r] - nm);
      const float p0 = __expf(s[0][r] - nm);
      const float p1 = __expf(s[1][r] - nm);
      s[0][r] = p0; s[1][r] = p1;
      float ts = p0 + p1;
#pragma unroll
      for (int o = 8; o >= 1; o >>= 1) ts += __shfl_xor(ts, o);
      rsum[r] = rsum[r] * corr + ts;
      rmax[r] = nm;
      acc[0][r] *= corr; acc[1][r] *= corr; acc[2][r] *= corr; acc[3][r] *= corr;
    }

    // ---- P (D-layout) -> LDS as 16x32 row-major bf16 ----
#pragma unroll
    for (int j = 0; j < 2; ++j) {
#pragma unroll
      for (int r = 0; r < 8; ++r)
        sP[wave][(r + 8 * half) * 32 + 16 * j + nl] = f2bf(s[j][r]);
    }

    // ---- stage V tile transposed: sVt[dim][key] ----
    {
      const int key = k0 + lane;
      const int kc = key < SEQ_L ? key : SEQ_L - 1;
      union { uint4 q[8]; u16 u[64]; } vb;
#pragma unroll
      for (int i = 0; i < 8; ++i)
        vb.q[i] = *(const uint4*)&Vb[(size_t)kc * 64 + 8 * i];
#pragma unroll
      for (int dd = 0; dd < 64; ++dd)
        sVt[wave][dd * 32 + lane] = vb.u[dd];
    }
    asm volatile("s_wait_dscnt 0x0" ::: "memory");  // per-wave LDS RAW fence

    // ---- O += P(16x32) * V(32x64) ----
    FragAB fp;
    {
      const int klo = half ? 16 : 0;
      fp.q[0] = *(const uint4*)&sP[wave][nl * 32 + klo];
      fp.q[1] = *(const uint4*)&sP[wave][nl * 32 + 8 + klo];
    }
#pragma unroll
    for (int c = 0; c < 4; ++c) {
      FragAB fv;
      const int drow = 16 * c + nl;
      const int bk = half ? 16 : 0;
      fv.q[0] = *(const uint4*)&sVt[wave][drow * 32 + bk];
      fv.q[1] = *(const uint4*)&sVt[wave][drow * 32 + bk + 8];
      acc[c] = wmma_bf16(fp.v, fv.v, acc[c]);
    }
  }

  // ---- normalize and store as [B*L, H*64] bf16 ----
#pragma unroll
  for (int c = 0; c < 4; ++c) {
#pragma unroll
    for (int r = 0; r < 8; ++r) {
      const int m = r + 8 * half;
      const float o = acc[c][r] / rsum[r];
      outp[(size_t)(b * SEQ_L + q0 + m) * D_MODEL + h * 64 + 16 * c + nl] = f2bf(o);
    }
  }
}

// ---------------------------------------------------------------------------
// Launcher
// ---------------------------------------------------------------------------
extern "C" void kernel_launch(void* const* d_in, const int* in_sizes, int n_in,
                              void* d_out, int out_size, void* d_ws, size_t ws_size,
                              hipStream_t stream) {
  (void)in_sizes; (void)n_in; (void)out_size; (void)ws_size;
  const float* x      = (const float*)d_in[0];
  const unsigned char* kpm = (const unsigned char*)d_in[1];
  const float* qkv_w  = (const float*)d_in[2];
  const float* qkv_b  = (const float*)d_in[3];
  const float* proj_w = (const float*)d_in[4];
  const float* proj_b = (const float*)d_in[5];
  const float* ln1_g  = (const float*)d_in[6];
  const float* ln1_b  = (const float*)d_in[7];
  const float* ln2_g  = (const float*)d_in[8];
  const float* ln2_b  = (const float*)d_in[9];
  const float* w1     = (const float*)d_in[10];
  const float* w2     = (const float*)d_in[11];
  const float* w3     = (const float*)d_in[12];
  float* out = (float*)d_out;

  char* ws = (char*)d_ws;
  const size_t MB = (size_t)1 << 20;
  u16*   wqkv  = (u16*)(ws + 0 * MB);    // 6 MiB  [3072,1024] bf16
  u16*   wproj = (u16*)(ws + 6 * MB);    // 2 MiB  [1024,1024]
  u16*   w1b   = (u16*)(ws + 8 * MB);    // 8 MiB  [4096,1024]
  u16*   w2b   = (u16*)(ws + 16 * MB);   // 8 MiB
  u16*   w3b   = (u16*)(ws + 24 * MB);   // 8 MiB  [1024,4096]
  u16*   h1    = (u16*)(ws + 32 * MB);   // 8 MiB  LN1 out; reused as attn_out
  u16*   qkvb  = (u16*)(ws + 40 * MB);   // 24 MiB [4096,3072]
  u16*   qb    = (u16*)(ws + 64 * MB);   // 8 MiB  [B,H,L,64]
  u16*   kb    = (u16*)(ws + 72 * MB);   // 8 MiB
  u16*   vb    = (u16*)(ws + 80 * MB);   // 8 MiB
  float* x1    = (float*)(ws + 88 * MB); // 16 MiB residual-1 (f32)
  u16*   h2    = (u16*)(ws + 104 * MB);  // 8 MiB  LN2 out
  u16*   gate  = (u16*)(ws + 40 * MB);   // 32 MiB, overlays dead qkvb+qb

  // 1) weights -> bf16
  f32_to_bf16_kernel<<<(3 * D_MODEL * D_MODEL) / 256, 256, 0, stream>>>(qkv_w, wqkv, 3 * D_MODEL * D_MODEL);
  f32_to_bf16_kernel<<<(D_MODEL * D_MODEL) / 256, 256, 0, stream>>>(proj_w, wproj, D_MODEL * D_MODEL);
  f32_to_bf16_kernel<<<(D_FF * D_MODEL) / 256, 256, 0, stream>>>(w1, w1b, D_FF * D_MODEL);
  f32_to_bf16_kernel<<<(D_FF * D_MODEL) / 256, 256, 0, stream>>>(w2, w2b, D_FF * D_MODEL);
  f32_to_bf16_kernel<<<(D_MODEL * D_FF) / 256, 256, 0, stream>>>(w3, w3b, D_MODEL * D_FF);

  // 2) LN1
  ln_bf16_kernel<<<ROWS, 256, 0, stream>>>(x, ln1_g, ln1_b, h1);

  // 3) QKV GEMM -> bf16 [4096,3072]
  gemm_bf16_kernel<<<dim3((3 * D_MODEL) / 64, ROWS / 128), 256, 0, stream>>>(
      h1, wqkv, qkv_b, nullptr, nullptr, qkvb, ROWS, 3 * D_MODEL, D_MODEL);

  // 4) RoPE + head split
  rope_split_kernel<<<(BATCH * SEQ_L * N_HEADS * 32) / 256, 256, 0, stream>>>(qkvb, qb, kb, vb);

  // 5) Flash attention -> attn_out (reuses h1)
  attn_kernel<<<(BATCH * N_HEADS * (SEQ_L / 16)) / 8, 256, 0, stream>>>(qb, kb, vb, kpm, h1);

  // 6) proj GEMM + residual(x) -> x1 (f32)
  gemm_bf16_kernel<<<dim3(D_MODEL / 64, ROWS / 128), 256, 0, stream>>>(
      h1, wproj, proj_b, x, x1, nullptr, ROWS, D_MODEL, D_MODEL);

  // 7) LN2
  ln_bf16_kernel<<<ROWS, 256, 0, stream>>>(x1, ln2_g, ln2_b, h2);

  // 8) fused SwiGLU gate: silu(h2@w1^T)*(h2@w2^T) -> bf16 [4096,4096]
  mlp_gate_kernel<<<dim3(D_FF / 64, ROWS / 128), 256, 0, stream>>>(
      h2, w1b, w2b, gate, ROWS, D_FF, D_MODEL);

  // 9) w3 GEMM + residual(x1) -> d_out (f32)
  gemm_bf16_kernel<<<dim3(D_MODEL / 64, ROWS / 128), 256, 0, stream>>>(
      gate, w3b, nullptr, x1, out, nullptr, ROWS, D_MODEL, D_FF);
}
